// PhiMoESparseMoeBlock_78658031058967
// MI455X (gfx1250) — matrix operand order repaired and checked by reading, not compiled, not run on hardware
//
#include <hip/hip_runtime.h>
#include <cstdint>
#include <cstddef>

// ---------------------------------------------------------------------------
// PhiMoE sparse MoE block for gfx1250 (MI455X).
// Routed top-2 experts, bf16 WMMA (v_wmma_f32_16x16x32_bf16) with fp32 accum.
// M=32 token tiles; double-buffered LDS A staging via async direct-to-LDS
// loads (ASYNCcnt); K-loop unrolled 2x; B fragments software-pipelined
// (b_cur/b_next live together) to kill WMMA->VALU WAR hazard NOPs.
// ---------------------------------------------------------------------------

typedef __attribute__((ext_vector_type(16))) __bf16 v16bf;
typedef __attribute__((ext_vector_type(8)))  __bf16 v8bf;
typedef __attribute__((ext_vector_type(4)))  __bf16 v4bf;
typedef __attribute__((ext_vector_type(8)))  float  v8f;
typedef __attribute__((ext_vector_type(4)))  float  v4f;

constexpr int E = 8;
constexpr int H = 2048;
constexpr int F = 4096;
constexpr int T = 2048;          // B*S
constexpr float JITTER = 0.01f;

// ---- workspace layout (bytes) ---------------------------------------------
constexpr size_t WS_COUNTS = 0;                                   // E ints
constexpr size_t WS_OFFS   = 256;                                 // E ints
constexpr size_t WS_TOK    = 512;                                 // E*T ints
constexpr size_t WS_WT     = WS_TOK + sizeof(int) * (size_t)E * T;    // E*T floats
constexpr size_t WS_XB     = ((WS_WT + sizeof(float) * (size_t)E * T + 255) / 256) * 256; // T*H bf16
constexpr size_t WS_HBUF   = ((WS_XB + 2ull * T * H + 255) / 256) * 256;                  // 2T*F bf16
// total ~40.2 MB

// ---------------------------------------------------------------------------
// gfx1250 async direct-to-LDS helpers (CDNA5 ISA).
// ---------------------------------------------------------------------------
typedef __attribute__((address_space(3))) char lds_char;

__device__ __forceinline__ uint32_t to_lds(void* p) {
  return (uint32_t)(size_t)(lds_char*)p;        // generic -> LDS byte offset
}

__device__ __forceinline__ void async_load_b128(uint32_t lds, const void* g) {
  // LDS[lds] = MEM[g] (16 bytes), tracked with ASYNCcnt
  asm volatile("global_load_async_to_lds_b128 %0, %1, off"
               :: "v"(lds), "v"((uint64_t)(size_t)g) : "memory");
}

__device__ __forceinline__ void wait_async0() {
  asm volatile("s_wait_asynccnt 0" ::: "memory");
}

// ---------------------------------------------------------------------------
// Kernel 1: fp32 -> bf16 conversion of activations (done once; 8 MB).
// ---------------------------------------------------------------------------
__global__ void k_cvt_bf16(const float* __restrict__ x, __bf16* __restrict__ xb, int n4) {
  int i = blockIdx.x * blockDim.x + threadIdx.x;
  if (i < n4) {
    v4f f = reinterpret_cast<const v4f*>(x)[i];
    v4bf b;
    b.x = (__bf16)f.x; b.y = (__bf16)f.y; b.z = (__bf16)f.z; b.w = (__bf16)f.w;
    reinterpret_cast<v4bf*>(xb)[i] = b;
  }
}

// ---------------------------------------------------------------------------
// Kernel 2: router. scores = x @ gate_w^T (gate_w staged through LDS in 8KB
// chunks), masked-sampling top-2 (inference), scatter into per-expert lists.
// Grid is exactly T threads (8 x 256) -> no bounds divergence around barriers.
// ---------------------------------------------------------------------------
__global__ __launch_bounds__(256) void k_router(
    const float* __restrict__ x, const float* __restrict__ gw,
    int* __restrict__ counts, int* __restrict__ tok, float* __restrict__ wt) {
  constexpr int CH = 256;                        // H-chunk per stage
  __shared__ float sg[E * CH];                   // 8 KB

  const int t = blockIdx.x * 256 + threadIdx.x;  // == token id, always < T

  float s[E];
  #pragma unroll
  for (int e = 0; e < E; ++e) s[e] = 0.f;

  const v4f* xr = reinterpret_cast<const v4f*>(x + (size_t)t * H);

  for (int h0 = 0; h0 < H; h0 += CH) {
    #pragma unroll
    for (int i = 0; i < 2; ++i) {
      int u  = threadIdx.x + i * 256;            // float4 unit 0..511
      int e  = u >> 6;                           // 64 float4 per expert row
      int h4 = u & 63;
      reinterpret_cast<v4f*>(sg)[u] =
          reinterpret_cast<const v4f*>(gw + (size_t)e * H + h0)[h4];
    }
    __syncthreads();
    for (int h4 = 0; h4 < CH / 4; ++h4) {
      v4f xv = xr[(h0 >> 2) + h4];
      #pragma unroll
      for (int e = 0; e < E; ++e) {
        v4f g = reinterpret_cast<const v4f*>(sg)[e * (CH / 4) + h4];
        s[e] += xv.x * g.x + xv.y * g.y + xv.z * g.z + xv.w * g.w;
      }
    }
    __syncthreads();
  }

  // top-2 indices
  int i1 = 0;
  #pragma unroll
  for (int e = 1; e < E; ++e) if (s[e] > s[i1]) i1 = e;
  int i2 = (i1 == 0) ? 1 : 0;
  #pragma unroll
  for (int e = 0; e < E; ++e) if (e != i1 && s[e] > s[i2]) i2 = e;

  float v1 = s[i1], v2 = s[i2];
  float den1 = 0.f, den2 = 0.f;
  #pragma unroll
  for (int e = 0; e < E; ++e) {
    float a = fabsf(s[e]);
    bool drop1 = ((v1 - s[e]) / fmaxf(a, v1)) > 2.f * JITTER;
    if (!drop1) den1 += expf(s[e] - v1);
    bool drop2 = ((v2 - s[e]) / fmaxf(a, v2)) > 2.f * JITTER;
    if (!(drop2 || e == i1)) den2 += expf(s[e] - v2);
  }
  float m1 = 1.f / den1;   // p1[i1]: exp(v1-v1)=1 over kept sum
  float m2 = 1.f / den2;   // p2[i2]

  int p1 = atomicAdd(&counts[i1], 1);
  tok[i1 * T + p1] = t;  wt[i1 * T + p1] = m1;
  int p2 = atomicAdd(&counts[i2], 1);
  tok[i2 * T + p2] = t;  wt[i2 * T + p2] = m2;
}

// ---------------------------------------------------------------------------
// Kernel 3: tiny prefix sum over E experts (compact hbuf row offsets).
// ---------------------------------------------------------------------------
__global__ void k_offsets(const int* __restrict__ counts, int* __restrict__ offs) {
  if (blockIdx.x == 0 && threadIdx.x == 0) {
    int a = 0;
    for (int e = 0; e < E; ++e) { offs[e] = a; a += counts[e]; }
  }
}

// ---------------------------------------------------------------------------
// WMMA fragment helpers (A/B share a contiguous K order of this code's own; C/D per CDNA5 ISA §7.12.2).
// A: 16 consecutive K of row (lane&15); lanes>=16 take K+16.
// B: 16 consecutive K of column (lane&15); lanes>=16 take K+16.
// C/D: VGPR r -> row r (+8 for lanes>=16), col (lane&15).
// ---------------------------------------------------------------------------
__device__ __forceinline__ v16bf load_a_frag(const __bf16* rowBase, int pitch,
                                             int lane) {
  const int m  = lane & 15;
  const int c0 = (lane >= 16) ? 16 : 0;
  union { v16bf v; v8bf h[2]; } af;
  af.h[0] = *reinterpret_cast<const v8bf*>(rowBase + m * pitch + c0);
  af.h[1] = *reinterpret_cast<const v8bf*>(rowBase + m * pitch + c0 + 8);
  return af.v;
}

__device__ __forceinline__ v16bf cvt_b_frag(const float* p) {
  v16bf b;
  #pragma unroll
  for (int i = 0; i < 4; ++i) {
    v4f f = reinterpret_cast<const v4f*>(p)[i];
    b[4*i+0] = (__bf16)f.x; b[4*i+1] = (__bf16)f.y;
    b[4*i+2] = (__bf16)f.z; b[4*i+3] = (__bf16)f.w;
  }
  return b;
}

#define WMMA_BF16(A, B, C) \
  __builtin_amdgcn_wmma_f32_16x16x32_bf16(false, (A), false, (B), (short)0, (C), false, false)

constexpr int AP = 40;   // LDS A pitch (bf16 elems): conflict-free ds_read_b128

// ---------------------------------------------------------------------------
// Kernel 4: h = silu(x@w1_e^T) * (x@w3_e^T) for the expert's token list.
// Block = 128 threads (4 waves); tile = 32 tokens x 64 F-cols; K by 64 (2x32).
// B fragments for chunk k+32 are converted before the WMMAs of chunk k.
// ---------------------------------------------------------------------------
__global__ __launch_bounds__(128) void k_ffn1(
    const __bf16* __restrict__ xb, const float* __restrict__ w1,
    const float* __restrict__ w3, const int* __restrict__ counts,
    const int* __restrict__ offs, const int* __restrict__ tok,
    __bf16* __restrict__ hbuf) {
  const int e   = blockIdx.z;
  const int cnt = counts[e];
  const int m0  = blockIdx.y * 32;
  if (m0 >= cnt) return;                 // uniform early-exit
  const int n0  = blockIdx.x * 64;

  __shared__ __bf16 sA[2][32][AP];       // double-buffered 32x32 bf16 tile
  __shared__ int    sTok[32];

  const int tid = threadIdx.x;
  if (tid < 32) {
    int r = m0 + tid; if (r > cnt - 1) r = cnt - 1;
    sTok[tid] = tok[e * T + r];
  }
  __syncthreads();

  const int lane  = tid & 31;
  const int wid   = tid >> 5;
  const int col   = n0 + wid * 16 + (lane & 15);   // B column (= w1/w3 row)
  const int khalf = (lane >> 4) * 16;              // B K-range per half-wave

  const float* w1r = w1 + ((size_t)e * F + col) * H + khalf;
  const float* w3r = w3 + ((size_t)e * F + col) * H + khalf;

  const int arow = tid >> 2;            // 0..31
  const int acol = (tid & 3) * 8;       // 0,8,16,24
  const __bf16* aSrc = xb + (size_t)sTok[arow] * H + acol;
  const uint32_t ldsA0 = to_lds(&sA[0][arow][acol]);
  const uint32_t ldsA1 = to_lds(&sA[1][arow][acol]);

  v8f acc1[2] = {{}, {}};
  v8f acc3[2] = {{}, {}};

  // prologue: async-stage first A tile, convert first B fragments
  async_load_b128(ldsA0, aSrc);
  v16bf b1A = cvt_b_frag(w1r);
  v16bf b3A = cvt_b_frag(w3r);

  for (int k0 = 0; k0 < H; k0 += 64) {
    wait_async0();                      // buf0 tile resident in LDS
    __syncthreads();
    async_load_b128(ldsA1, aSrc + k0 + 32);     // k0+32 < H always
    __builtin_prefetch(w1r + k0 + 64, 0, 1);
    __builtin_prefetch(w3r + k0 + 64, 0, 1);
    // pipeline: convert NEXT chunk's B while consuming b?A (distinct regs)
    v16bf b1B = cvt_b_frag(w1r + k0 + 32);
    v16bf b3B = cvt_b_frag(w3r + k0 + 32);
    {
      v16bf a0 = load_a_frag(&sA[0][0][0],  AP, lane);
      v16bf a1 = load_a_frag(&sA[0][16][0], AP, lane);
      acc1[0] = WMMA_BF16(a0, b1A, acc1[0]);
      acc1[1] = WMMA_BF16(a1, b1A, acc1[1]);
      acc3[0] = WMMA_BF16(a0, b3A, acc3[0]);
      acc3[1] = WMMA_BF16(a1, b3A, acc3[1]);
    }
    wait_async0();                      // buf1 tile resident in LDS
    __syncthreads();
    if (k0 + 64 < H) {
      async_load_b128(ldsA0, aSrc + k0 + 64);
      b1A = cvt_b_frag(w1r + k0 + 64);
      b3A = cvt_b_frag(w3r + k0 + 64);
      __builtin_prefetch(w1r + k0 + 96, 0, 1);
      __builtin_prefetch(w3r + k0 + 96, 0, 1);
    }
    {
      v16bf a0 = load_a_frag(&sA[1][0][0],  AP, lane);
      v16bf a1 = load_a_frag(&sA[1][16][0], AP, lane);
      acc1[0] = WMMA_BF16(a0, b1B, acc1[0]);
      acc1[1] = WMMA_BF16(a1, b1B, acc1[1]);
      acc3[0] = WMMA_BF16(a0, b3B, acc3[0]);
      acc3[1] = WMMA_BF16(a1, b3B, acc3[1]);
    }
  }

  // epilogue: h = silu(acc1) * acc3 -> bf16 hbuf (compact rows)
  const int nloc  = lane & 15;
  const int rbase = (lane >> 4) * 8;
  const int hrow0 = offs[e];
  #pragma unroll
  for (int sub = 0; sub < 2; ++sub) {
    #pragma unroll
    for (int r = 0; r < 8; ++r) {
      int row  = sub * 16 + rbase + r;
      int grow = m0 + row;
      if (grow < cnt) {
        float g1 = acc1[sub][r], g3 = acc3[sub][r];
        float sig = __builtin_amdgcn_rcpf(1.f + __expf(-g1));   // v_rcp_f32
        hbuf[(size_t)(hrow0 + grow) * F + n0 + wid * 16 + nloc] =
            (__bf16)(g1 * sig * g3);
      }
    }
  }
}

// ---------------------------------------------------------------------------
// Kernel 5: y = h @ w2_e^T, scaled scatter-add into out via f32 atomics.
// Block = 128 threads; tile = 32 rows x 64 H-cols; K over F by 64 (2x32).
// ---------------------------------------------------------------------------
__global__ __launch_bounds__(128) void k_ffn2(
    const __bf16* __restrict__ hbuf, const float* __restrict__ w2,
    const int* __restrict__ counts, const int* __restrict__ offs,
    const int* __restrict__ tok, const float* __restrict__ wt,
    float* __restrict__ out) {
  const int e   = blockIdx.z;
  const int cnt = counts[e];
  const int m0  = blockIdx.y * 32;
  if (m0 >= cnt) return;
  const int n0   = blockIdx.x * 64;
  const int base = offs[e];

  __shared__ __bf16 sA[2][32][AP];
  __shared__ int    sTok[32];
  __shared__ float  sW[32];

  const int tid = threadIdx.x;
  if (tid < 32) {
    int r = m0 + tid; if (r > cnt - 1) r = cnt - 1;
    sTok[tid] = tok[e * T + r];
    sW[tid]   = wt[e * T + r];
  }
  __syncthreads();

  const int lane  = tid & 31;
  const int wid   = tid >> 5;
  const int col   = n0 + wid * 16 + (lane & 15);   // output H column (= w2 row)
  const int khalf = (lane >> 4) * 16;
  const float* w2r = w2 + ((size_t)e * H + col) * F + khalf;

  const int arow = tid >> 2;
  const int acol = (tid & 3) * 8;
  int rloc = m0 + arow; if (rloc > cnt - 1) rloc = cnt - 1;
  const __bf16* aSrc = hbuf + (size_t)(base + rloc) * F + acol;
  const uint32_t ldsA0 = to_lds(&sA[0][arow][acol]);
  const uint32_t ldsA1 = to_lds(&sA[1][arow][acol]);

  v8f acc[2] = {{}, {}};

  async_load_b128(ldsA0, aSrc);
  v16bf bA = cvt_b_frag(w2r);

  for (int k0 = 0; k0 < F; k0 += 64) {
    wait_async0();
    __syncthreads();
    async_load_b128(ldsA1, aSrc + k0 + 32);
    __builtin_prefetch(w2r + k0 + 64, 0, 1);
    v16bf bB = cvt_b_frag(w2r + k0 + 32);
    {
      v16bf a0 = load_a_frag(&sA[0][0][0],  AP, lane);
      v16bf a1 = load_a_frag(&sA[0][16][0], AP, lane);
      acc[0] = WMMA_BF16(a0, bA, acc[0]);
      acc[1] = WMMA_BF16(a1, bA, acc[1]);
    }
    wait_async0();
    __syncthreads();
    if (k0 + 64 < F) {
      async_load_b128(ldsA0, aSrc + k0 + 64);
      bA = cvt_b_frag(w2r + k0 + 64);
      __builtin_prefetch(w2r + k0 + 96, 0, 1);
    }
    {
      v16bf a0 = load_a_frag(&sA[1][0][0],  AP, lane);
      v16bf a1 = load_a_frag(&sA[1][16][0], AP, lane);
      acc[0] = WMMA_BF16(a0, bB, acc[0]);
      acc[1] = WMMA_BF16(a1, bB, acc[1]);
    }
  }

  const int nloc  = lane & 15;
  const int rbase = (lane >> 4) * 8;
  #pragma unroll
  for (int sub = 0; sub < 2; ++sub) {
    #pragma unroll
    for (int r = 0; r < 8; ++r) {
      int row  = sub * 16 + rbase + r;
      int grow = m0 + row;
      if (grow < cnt) {
        int   t = sTok[row];
        float w = sW[row];
        atomicAdd(&out[(size_t)t * H + n0 + wid * 16 + nloc], w * acc[sub][r]);
      }
    }
  }
}

// ---------------------------------------------------------------------------
extern "C" void kernel_launch(void* const* d_in, const int* in_sizes, int n_in,
                              void* d_out, int out_size, void* d_ws, size_t ws_size,
                              hipStream_t stream) {
  const float* x  = (const float*)d_in[0];  // [B,S,H]
  const float* gw = (const float*)d_in[1];  // [E,H]
  const float* w1 = (const float*)d_in[2];  // [E,F,H]
  const float* w2 = (const float*)d_in[3];  // [E,H,F]
  const float* w3 = (const float*)d_in[4];  // [E,F,H]
  float* out = (float*)d_out;

  char* ws = (char*)d_ws;
  int*    counts = (int*)(ws + WS_COUNTS);
  int*    offs   = (int*)(ws + WS_OFFS);
  int*    tokl   = (int*)(ws + WS_TOK);
  float*  wtl    = (float*)(ws + WS_WT);
  __bf16* xb     = (__bf16*)(ws + WS_XB);
  __bf16* hbuf   = (__bf16*)(ws + WS_HBUF);

  hipMemsetAsync(counts, 0, sizeof(int) * E, stream);
  hipMemsetAsync(out, 0, sizeof(float) * (size_t)T * H, stream);

  k_cvt_bf16<<<(T * H / 4 + 255) / 256, 256, 0, stream>>>(x, xb, T * H / 4);
  k_router<<<T / 256, 256, 0, stream>>>(x, gw, counts, tokl, wtl);
  k_offsets<<<1, 1, 0, stream>>>(counts, offs);

  dim3 g1(F / 64, T / 32, E);
  k_ffn1<<<g1, 128, 0, stream>>>(xb, w1, w3, counts, offs, tokl, hbuf);

  dim3 g2(H / 64, T / 32, E);
  k_ffn2<<<g2, 128, 0, stream>>>(hbuf, w2, counts, offs, tokl, wtl, out);
}
